// LSTMSeq2SeqDifferent_54322746360201
// MI455X (gfx1250) — compile-verified, matrix operand order checked by reading, not run on hardware
//
#include <hip/hip_runtime.h>
#include <hip/hip_bf16.h>
#include <math.h>

// ---------------------------------------------------------------------------
// LSTM seq2seq on gfx1250 (MI455X): bf16 WMMA for all GEMMs.
// T=512, B=64, D_IN=256, H=512, D_OUT=256, 4H=2048.
// ---------------------------------------------------------------------------

typedef __bf16 bf16_t;
typedef __attribute__((ext_vector_type(8)))  bf16_t v8bf;
typedef __attribute__((ext_vector_type(16))) bf16_t v16bf;
typedef __attribute__((ext_vector_type(8)))  float  v8f;

#define TSTEPS 512
#define BATCH  64
#define DIN    256
#define HID    512
#define DOUT   256
#define G4H    2048
#define NWG    16

// Load 16 bf16 elements for the A or B operand of v_wmma_f32_16x16x32_bf16.
// Per ISA layout (16-bit A 16x32): lane<16 holds k = kk+{0..7, 16..23},
// lane>=16 holds k = kk+{8..15, 24..31}.  B (32x16) is the mirror layout with
// the lane indexing the N column, i.e. a row of W (since B = W^T).
__device__ __forceinline__ v16bf load_ab(const bf16_t* __restrict__ rowbase,
                                         int kk, int off0) {
  v8bf lo = *reinterpret_cast<const v8bf*>(rowbase + kk + off0);
  v8bf hi = *reinterpret_cast<const v8bf*>(rowbase + kk + 16 + off0);
  return __builtin_shufflevector(lo, hi, 0, 1, 2, 3, 4, 5, 6, 7,
                                         8, 9, 10, 11, 12, 13, 14, 15);
}

__device__ __forceinline__ v8f wmma_bf16(v16bf a, v16bf b, v8f c) {
  return __builtin_amdgcn_wmma_f32_16x16x32_bf16(
      /*neg_a=*/false, a, /*neg_b=*/false, b,
      /*c_mod=*/(short)0, c, /*reuse_a=*/false, /*reuse_b=*/false);
}

__device__ __forceinline__ float sigmoidf_(float x) {
  return 1.0f / (1.0f + expf(-x));
}

// ---------------------------------------------------------------------------
// Prep kernels
// ---------------------------------------------------------------------------
__global__ void k_f32_to_bf16(const float* __restrict__ src,
                              bf16_t* __restrict__ dst, int n) {
  int i = blockIdx.x * blockDim.x + threadIdx.x;
  if (i < n) dst[i] = (bf16_t)src[i];
}

// dec_in[0]=0, dec_in[t]=target[t-1]  (flat over T*B*DOUT)
__global__ void k_build_dec(const float* __restrict__ target,
                            bf16_t* __restrict__ dst) {
  int i = blockIdx.x * blockDim.x + threadIdx.x;
  const int n = TSTEPS * BATCH * DOUT;
  if (i < n)
    dst[i] = (i < BATCH * DOUT) ? (bf16_t)0.0f : (bf16_t)target[i - BATCH * DOUT];
}

__global__ void k_zero_bar(unsigned* __restrict__ bar) {
  if (threadIdx.x == 0 && blockIdx.x == 0) *bar = 0u;
}

// ---------------------------------------------------------------------------
// Persistent cooperative scan kernel: 1024 steps (512 encoder + 512 decoder).
// 16 WGs x 256 threads. WG w owns hidden units [32w, 32w+32) for all 4 gates.
// One grid-wide sync (atomic counter) per step; h double-buffered in global.
// c state lives in registers for the whole scan.
// ---------------------------------------------------------------------------
__global__ void __launch_bounds__(256, 1)
k_scan(const bf16_t* __restrict__ x_bf, const bf16_t* __restrict__ dec_bf,
       const bf16_t* __restrict__ eWih, const bf16_t* __restrict__ eWhh,
       const bf16_t* __restrict__ dWih, const bf16_t* __restrict__ dWhh,
       const float* __restrict__ enc_b, const float* __restrict__ dec_b,
       const float* __restrict__ cell0,
       bf16_t* __restrict__ h_bf,        // 2 * BATCH*HID (double buffered)
       bf16_t* __restrict__ hs_bf,       // TSTEPS * BATCH*HID (decoder h's)
       unsigned* __restrict__ bar) {
  __shared__ float gate_lds[4][BATCH][32];   // activated i,f,g,o for this chunk

  const int tid  = threadIdx.x;
  const int wave = tid >> 5;
  const int lane = tid & 31;
  const int w    = blockIdx.x;                       // hidden chunk id
  const int gidx = wave >> 1;                        // 0:i 1:f 2:g 3:o
  const int lcol = ((wave & 1) << 4) + (lane & 15);  // local col 0..31
  const int gcol = HID * gidx + 32 * w + lcol;       // column in 4H
  const int off0 = (lane < 16) ? 0 : 8;
  const int mrow = lane & 15;
  const int rbase = (lane < 16) ? 0 : 8;

  // elementwise-update assignment: col = tid%32, rows r0..r0+7
  const int ucol = tid & 31;
  const int r0   = (tid >> 5) * 8;
  float c_reg[8];
#pragma unroll
  for (int q = 0; q < 8; ++q)
    c_reg[q] = cell0[(size_t)(r0 + q) * HID + 32 * w + ucol];

  for (int s = 0; s < 2 * TSTEPS; ++s) {
    const bool enc = (s < TSTEPS);
    const bf16_t* hcur = h_bf + (size_t)(s & 1) * (BATCH * HID);
    const bf16_t* Whh_row = (enc ? eWhh : dWhh) + (size_t)gcol * HID;
    const bf16_t* Wih_row = (enc ? eWih : dWih) + (size_t)gcol * DIN;
    const bf16_t* xt = enc ? (x_bf + (size_t)s * (BATCH * DIN))
                           : (dec_bf + (size_t)(s - TSTEPS) * (BATCH * DIN));
    const float bias = enc ? enc_b[gcol] : dec_b[gcol];

    v8f acc[4];
#pragma unroll
    for (int m = 0; m < 4; ++m)
#pragma unroll
      for (int r = 0; r < 8; ++r) acc[m][r] = bias;

    // gates += h @ Whh^T   (K = 512)
    for (int kk = 0; kk < HID; kk += 32) {
      v16bf bmat = load_ab(Whh_row, kk, off0);
#pragma unroll
      for (int m = 0; m < 4; ++m) {
        v16bf amat = load_ab(hcur + (size_t)(16 * m + mrow) * HID, kk, off0);
        acc[m] = wmma_bf16(amat, bmat, acc[m]);
      }
    }
    // gates += x_t @ Wih^T  (K = 256)
    for (int kk = 0; kk < DIN; kk += 32) {
      v16bf bmat = load_ab(Wih_row, kk, off0);
#pragma unroll
      for (int m = 0; m < 4; ++m) {
        v16bf amat = load_ab(xt + (size_t)(16 * m + mrow) * DIN, kk, off0);
        acc[m] = wmma_bf16(amat, bmat, acc[m]);
      }
    }

    // activations -> LDS  (C/D layout: elem r -> row 16m + rbase + r, col lane%16)
#pragma unroll
    for (int m = 0; m < 4; ++m)
#pragma unroll
      for (int r = 0; r < 8; ++r) {
        float v = acc[m][r];
        float a = (gidx == 2) ? tanhf(v) : sigmoidf_(v);
        gate_lds[gidx][16 * m + rbase + r][lcol] = a;
      }
    __syncthreads();

    // c/h elementwise update for this WG's hidden slice
    bf16_t* hnext = h_bf + (size_t)((s + 1) & 1) * (BATCH * HID);
    bf16_t* hsrow = enc ? nullptr
                        : hs_bf + (size_t)(s - TSTEPS) * (BATCH * HID);
#pragma unroll
    for (int q = 0; q < 8; ++q) {
      int r = r0 + q;
      float iv = gate_lds[0][r][ucol];
      float fv = gate_lds[1][r][ucol];
      float gv = gate_lds[2][r][ucol];
      float ov = gate_lds[3][r][ucol];
      float c = fv * c_reg[q] + iv * gv;
      c_reg[q] = c;
      float h = ov * tanhf(c);
      bf16_t hb = (bf16_t)h;
      hnext[(size_t)r * HID + 32 * w + ucol] = hb;
      if (!enc) hsrow[(size_t)r * HID + 32 * w + ucol] = hb;
    }
    __syncthreads();  // gate_lds reused next step

    // grid-wide release/acquire sync
    if (tid == 0) {
      __threadfence();
      atomicAdd(bar, 1u);
      const unsigned tgt = (unsigned)NWG * (unsigned)(s + 1);
      while (atomicAdd(bar, 0u) < tgt) { __builtin_amdgcn_s_sleep(1); }
      __threadfence();
    }
    __syncthreads();
  }
}

// ---------------------------------------------------------------------------
// FC: logits[M=32768, N=256] = hs[M,512] @ fc_W^T + fc_b   (bf16 WMMA)
// One wave = 16x64 strip (1 M-tile x 4 N-tiles, A reused 4x).
// ---------------------------------------------------------------------------
__global__ void __launch_bounds__(256)
k_fc(const bf16_t* __restrict__ hs, const bf16_t* __restrict__ fcW,
     const float* __restrict__ fcb, float* __restrict__ out) {
  const int tid  = threadIdx.x;
  const int lane = tid & 31;
  const int wave = tid >> 5;
  const int unit = blockIdx.x * 8 + wave;  // 8192 wave-units
  const int mt   = unit >> 2;              // 0..2047
  const int nb   = (unit & 3) * 64;        // N base
  const int off0 = (lane < 16) ? 0 : 8;
  const int mrow = lane & 15;
  const int rbase = (lane < 16) ? 0 : 8;

  v8f acc[4];
#pragma unroll
  for (int j = 0; j < 4; ++j) {
    float b = fcb[nb + 16 * j + (lane & 15)];
#pragma unroll
    for (int r = 0; r < 8; ++r) acc[j][r] = b;
  }

  for (int kk = 0; kk < HID; kk += 32) {
    v16bf amat = load_ab(hs + (size_t)(16 * mt + mrow) * HID, kk, off0);
#pragma unroll
    for (int j = 0; j < 4; ++j) {
      v16bf bmat = load_ab(fcW + (size_t)(nb + 16 * j + mrow) * HID, kk, off0);
      acc[j] = wmma_bf16(amat, bmat, acc[j]);
    }
  }

#pragma unroll
  for (int j = 0; j < 4; ++j)
#pragma unroll
    for (int r = 0; r < 8; ++r) {
      int row = 16 * mt + rbase + r;
      int col = nb + 16 * j + (lane & 15);
      out[(size_t)row * DOUT + col] = acc[j][r];
    }
}

// ---------------------------------------------------------------------------
// In-place row softmax over 256 cols; one block per row.
// ---------------------------------------------------------------------------
__global__ void __launch_bounds__(256)
k_softmax(float* __restrict__ out) {
  __shared__ float red[256];
  const int row = blockIdx.x;
  const int tid = threadIdx.x;
  float v = out[(size_t)row * DOUT + tid];
  red[tid] = v;
  __syncthreads();
  for (int s = 128; s > 0; s >>= 1) {
    if (tid < s) red[tid] = fmaxf(red[tid], red[tid + s]);
    __syncthreads();
  }
  float mx = red[0];
  __syncthreads();
  float e = expf(v - mx);
  red[tid] = e;
  __syncthreads();
  for (int s = 128; s > 0; s >>= 1) {
    if (tid < s) red[tid] += red[tid + s];
    __syncthreads();
  }
  float sum = red[0];
  out[(size_t)row * DOUT + tid] = e / sum;
}

// ---------------------------------------------------------------------------
// Host launcher
// ---------------------------------------------------------------------------
extern "C" void kernel_launch(void* const* d_in, const int* in_sizes, int n_in,
                              void* d_out, int out_size, void* d_ws,
                              size_t ws_size, hipStream_t stream) {
  (void)in_sizes; (void)n_in; (void)out_size; (void)ws_size;

  const float* x      = (const float*)d_in[0];   // [T,B,DIN]
  const float* target = (const float*)d_in[1];   // [T,B,DOUT]
  const float* h0     = (const float*)d_in[2];   // [1,B,H]
  const float* c0     = (const float*)d_in[3];   // [1,B,H]
  const float* eWih   = (const float*)d_in[4];   // [4H,DIN]
  const float* eWhh   = (const float*)d_in[5];   // [4H,H]
  const float* eb     = (const float*)d_in[6];   // [4H]
  const float* dWih   = (const float*)d_in[7];   // [4H,DOUT]
  const float* dWhh   = (const float*)d_in[8];   // [4H,H]
  const float* db     = (const float*)d_in[9];   // [4H]
  const float* fcW    = (const float*)d_in[10];  // [DOUT,H]
  const float* fcb    = (const float*)d_in[11];  // [DOUT]
  float* out = (float*)d_out;

  // workspace carve-out (256B aligned slabs)
  char* ws = (char*)d_ws;
  size_t off = 0;
  auto carve = [&](size_t bytes) -> char* {
    char* p = ws + off;
    off = (off + bytes + 255) & ~(size_t)255;
    return p;
  };
  unsigned* bar     = (unsigned*)carve(256);
  bf16_t* x_bf      = (bf16_t*)carve((size_t)TSTEPS * BATCH * DIN * 2);
  bf16_t* dec_bf    = (bf16_t*)carve((size_t)TSTEPS * BATCH * DOUT * 2);
  bf16_t* eWih_bf   = (bf16_t*)carve((size_t)G4H * DIN * 2);
  bf16_t* eWhh_bf   = (bf16_t*)carve((size_t)G4H * HID * 2);
  bf16_t* dWih_bf   = (bf16_t*)carve((size_t)G4H * DOUT * 2);
  bf16_t* dWhh_bf   = (bf16_t*)carve((size_t)G4H * HID * 2);
  bf16_t* fcW_bf    = (bf16_t*)carve((size_t)DOUT * HID * 2);
  bf16_t* h_bf      = (bf16_t*)carve((size_t)2 * BATCH * HID * 2);
  bf16_t* hs_bf     = (bf16_t*)carve((size_t)TSTEPS * BATCH * HID * 2);

  const int cvt_threads = 256;
  auto blocks = [](int n, int t) { return (n + t - 1) / t; };

  k_zero_bar<<<1, 32, 0, stream>>>(bar);

  const int n_x = TSTEPS * BATCH * DIN;
  k_f32_to_bf16<<<blocks(n_x, cvt_threads), cvt_threads, 0, stream>>>(x, x_bf, n_x);
  k_build_dec<<<blocks(TSTEPS * BATCH * DOUT, cvt_threads), cvt_threads, 0, stream>>>(target, dec_bf);
  k_f32_to_bf16<<<blocks(G4H * DIN, cvt_threads), cvt_threads, 0, stream>>>(eWih, eWih_bf, G4H * DIN);
  k_f32_to_bf16<<<blocks(G4H * HID, cvt_threads), cvt_threads, 0, stream>>>(eWhh, eWhh_bf, G4H * HID);
  k_f32_to_bf16<<<blocks(G4H * DOUT, cvt_threads), cvt_threads, 0, stream>>>(dWih, dWih_bf, G4H * DOUT);
  k_f32_to_bf16<<<blocks(G4H * HID, cvt_threads), cvt_threads, 0, stream>>>(dWhh, dWhh_bf, G4H * HID);
  k_f32_to_bf16<<<blocks(DOUT * HID, cvt_threads), cvt_threads, 0, stream>>>(fcW, fcW_bf, DOUT * HID);
  // initial h into buffer slot 0
  k_f32_to_bf16<<<blocks(BATCH * HID, cvt_threads), cvt_threads, 0, stream>>>(h0, h_bf, BATCH * HID);

  // the whole recurrence in one persistent cooperative launch
  k_scan<<<NWG, 256, 0, stream>>>(x_bf, dec_bf, eWih_bf, eWhh_bf, dWih_bf,
                                  dWhh_bf, eb, db, c0, h_bf, hs_bf, bar);

  // FC + softmax epilogue
  k_fc<<<(TSTEPS * BATCH / 16) * 4 / 8, 256, 0, stream>>>(hs_bf, fcW_bf, fcb, out);
  k_softmax<<<TSTEPS * BATCH, 256, 0, stream>>>(out);
}
